// EncoderBlock_82317343195502
// MI455X (gfx1250) — compile-verified, matrix operand order checked by reading, not run on hardware
//
#include <hip/hip_runtime.h>
#include <math.h>

// ---------------------------------------------------------------------------
// Types for CDNA5 WMMA (gfx1250, wave32)
// ---------------------------------------------------------------------------
typedef __bf16 bf16_t;
typedef __bf16 v16bf __attribute__((ext_vector_type(16)));
typedef float  v8f   __attribute__((ext_vector_type(8)));
typedef int    v4i_b128 __attribute__((vector_size(16)));   // matches builtin param

// Async global->LDS copy path (gfx1250): GLOBAL_LOAD_ASYNC_TO_LDS_B128,
// tracked by ASYNCcnt. Guarded so we degrade to plain LDS staging if the
// toolchain lacks the builtin.
#if defined(__AMDGCN__) && defined(__gfx1250__) && \
    __has_builtin(__builtin_amdgcn_global_load_async_to_lds_b128)
#define HAVE_ASYNC_LDS 1
#else
#define HAVE_ASYNC_LDS 0
#endif

__device__ __forceinline__ void async_copy16(void* lds, const void* g) {
#if HAVE_ASYNC_LDS
    __builtin_amdgcn_global_load_async_to_lds_b128(
        (v4i_b128*)g, (v4i_b128*)lds, 0, 0);
#else
    *(uint4*)lds = *(const uint4*)g;
#endif
}
__device__ __forceinline__ void async_wait0() {
#if HAVE_ASYNC_LDS
    asm volatile("s_wait_asynccnt 0x0" ::: "memory");
#endif
}

// float -> bf16 round-to-nearest-even
__device__ __forceinline__ unsigned short f2bf_bits(float f) {
    union { float f; unsigned u; } x; x.f = f;
    return (unsigned short)((x.u + 0x7FFFu + ((x.u >> 16) & 1u)) >> 16);
}
__device__ __forceinline__ bf16_t f2bf(float f) {
    union { unsigned short s; __bf16 b; } y; y.s = f2bf_bits(f);
    return y.b;
}

__device__ __forceinline__ v8f wmma_bf16(v16bf a, v16bf b, v8f c) {
    // (neg_a, A, neg_b, B, c_mod, C, reuse_a, reuse_b)
    return __builtin_amdgcn_wmma_f32_16x16x32_bf16(false, a, false, b, (short)0, c, false, false);
}

// Load one 16x32 bf16 fragment from LDS following the documented A/B layout:
// lane<16 : row=row0+lane,    elems = K[0..7],  K[16..23]
// lane>=16: row=row0+lane-16, elems = K[8..15], K[24..31]
union FragU { v16bf v; uint4 u[2]; };
__device__ __forceinline__ v16bf load_frag(const bf16_t* base, int row0, int stride, int lane) {
    int half = (lane >> 4) & 1;
    int r    = row0 + (lane & 15);
    const bf16_t* p = base + r * stride + half * 8;
    FragU f;
    f.u[0] = *(const uint4*)(p);
    f.u[1] = *(const uint4*)(p + 16);
    return f.v;
}

// 16-lane (half-wave) reductions — wave32, xor masks < 16 stay in-group
__device__ __forceinline__ float redmax16(float v) {
#pragma unroll
    for (int off = 1; off < 16; off <<= 1) v = fmaxf(v, __shfl_xor(v, off, 32));
    return v;
}
__device__ __forceinline__ float redsum16(float v) {
#pragma unroll
    for (int off = 1; off < 16; off <<= 1) v += __shfl_xor(v, off, 32);
    return v;
}

// ---------------------------------------------------------------------------
// fp32 -> bf16 convert, 4 elems / thread (n must be a multiple of 4 — true here)
// ---------------------------------------------------------------------------
__global__ void cvt_bf16_kernel(const float4* __restrict__ in, uint2* __restrict__ out, int n4) {
    int i = blockIdx.x * blockDim.x + threadIdx.x;
    if (i < n4) {
        float4 v = in[i];
        uint2 o;
        o.x = (unsigned)f2bf_bits(v.x) | ((unsigned)f2bf_bits(v.y) << 16);
        o.y = (unsigned)f2bf_bits(v.z) | ((unsigned)f2bf_bits(v.w) << 16);
        out[i] = o;
    }
}

// ---------------------------------------------------------------------------
// GEMM: C[M,N] = A[M,K](bf16) @ W[N,K]^T(bf16) + bias[N], optional relu.
// Block 256 thr = 8 waves; 128x128 tile; BK=32 (one WMMA K-step).
// Each wave: 64x32 sub-tile = 4x2 v8f accumulators.
// Double-buffered LDS: async DMA of tile k+1 overlaps WMMAs of tile k.
// ---------------------------------------------------------------------------
#define GBM 128
#define GBN 128
#define GBK 32
#define GLD 40   // padded LDS stride (elems)

template <int ACT, bool WF, bool WB>
__global__ __launch_bounds__(256)
void gemm_bf16_kernel(const bf16_t* __restrict__ A, const bf16_t* __restrict__ W,
                      const float* __restrict__ bias,
                      float* __restrict__ outF, bf16_t* __restrict__ outB,
                      int M, int N, int K)
{
    __shared__ bf16_t sA[2][GBM * GLD];
    __shared__ bf16_t sB[2][GBN * GLD];

    const int tid  = threadIdx.x;
    const int lane = tid & 31;
    const int w    = tid >> 5;        // 0..7
    const int wm   = (w >> 2) * 64;   // 0,64
    const int wn   = (w & 3) * 32;    // 0,32,64,96
    const int bm   = blockIdx.y * GBM;
    const int bn   = blockIdx.x * GBN;
    const int half = lane >> 4, l16 = lane & 15;

    // bias: each lane only ever touches two output columns
    float biasr[2];
    biasr[0] = bias[bn + wn + l16];
    biasr[1] = bias[bn + wn + 16 + l16];

    v8f acc[4][2];
#pragma unroll
    for (int i = 0; i < 4; ++i)
#pragma unroll
        for (int j = 0; j < 2; ++j) acc[i][j] = (v8f){0,0,0,0,0,0,0,0};

    const int lrow = tid >> 1;          // 0..127
    const int lcol = (tid & 1) * 16;    // 0,16
    const bf16_t* gA = A + (size_t)(bm + lrow) * K + lcol;
    const bf16_t* gW = W + (size_t)(bn + lrow) * K + lcol;
    const int lofs = lrow * GLD + lcol;

    // prologue: stage tile 0 into buffer 0
    async_copy16(&sA[0][lofs],     gA);
    async_copy16(&sA[0][lofs + 8], gA + 8);
    async_copy16(&sB[0][lofs],     gW);
    async_copy16(&sB[0][lofs + 8], gW + 8);
    async_wait0();
    __syncthreads();

    const int nsteps = K / GBK;
    for (int s = 0; s < nsteps; ++s) {
        const int cur = s & 1;
        // issue next tile's async DMA before computing — overlaps with WMMA
        if (s + 1 < nsteps) {
            const int nxt = cur ^ 1, kk = (s + 1) * GBK;
            async_copy16(&sA[nxt][lofs],     gA + kk);
            async_copy16(&sA[nxt][lofs + 8], gA + kk + 8);
            async_copy16(&sB[nxt][lofs],     gW + kk);
            async_copy16(&sB[nxt][lofs + 8], gW + kk + 8);
        }

        v16bf af[4], bfr[2];
#pragma unroll
        for (int i = 0; i < 4; ++i) af[i]  = load_frag(sA[cur], wm + 16 * i, GLD, lane);
#pragma unroll
        for (int j = 0; j < 2; ++j) bfr[j] = load_frag(sB[cur], wn + 16 * j, GLD, lane);
#pragma unroll
        for (int i = 0; i < 4; ++i)
#pragma unroll
            for (int j = 0; j < 2; ++j)
                acc[i][j] = wmma_bf16(af[i], bfr[j], acc[i][j]);

        async_wait0();      // next buffer's DMA has landed (ran under the WMMAs)
        __syncthreads();
    }

    // Epilogue. C layout: row = r + 8*(lane>=16), col = lane&15 within 16x16.
#pragma unroll
    for (int i = 0; i < 4; ++i)
#pragma unroll
        for (int j = 0; j < 2; ++j)
#pragma unroll
            for (int r = 0; r < 8; ++r) {
                int row = bm + wm + 16 * i + r + 8 * half;
                int col = bn + wn + 16 * j + l16;
                float v = acc[i][j][r] + biasr[j];
                if (ACT) v = fmaxf(v, 0.0f);
                size_t idx = (size_t)row * N + col;
                if (WF) outF[idx] = v;
                if (WB) outB[idx] = f2bf(v);
            }
}

// ---------------------------------------------------------------------------
// Fused flash-style attention for qm/km/vm viewed as [64][1024][64] bf16.
// Grid: (L/64 query tiles, B*H). Block: 128 thr = 4 waves, wave owns 16 rows.
// ---------------------------------------------------------------------------
#define AL   1024
#define ADH  64
#define ASP  72   // padded LDS stride

__device__ __forceinline__ int rel_bucket(int qi, int ki) {
    int n = ki - qi;                 // n = -(q - k)
    int ret = (n < 0) ? 16 : 0;
    n = n < 0 ? -n : n;
    int v;
    if (n < 8) v = n;
    else {
        v = 8 + (int)(__logf((float)n * 0.125f) * (8.0f / __logf(16.0f)));
        v = v < 15 ? v : 15;
    }
    return ret + v;
}

__global__ __launch_bounds__(128)
void flash_attn_kernel(const bf16_t* __restrict__ qb, const bf16_t* __restrict__ kb,
                       const bf16_t* __restrict__ vb, const float* __restrict__ rel_emb,
                       const int* __restrict__ pmask, bf16_t* __restrict__ attb)
{
    __shared__ bf16_t sQ [64 * ASP];
    __shared__ bf16_t sK [64 * ASP];
    __shared__ bf16_t sVt[64 * ASP];   // transposed: [dh][key]
    __shared__ bf16_t sP [64 * ASP];

    const int qt   = blockIdx.x;       // 64-row query tile
    const int bh   = blockIdx.y;       // 0..63  (B*H)
    const int tid  = threadIdx.x;
    const int lane = tid & 31;
    const int w    = tid >> 5;         // 0..3
    const int half = lane >> 4, l16 = lane & 15;
    const int bidx = bh >> 4;          // batch
    const int head = bh & 15;

    const int srow = tid >> 1, sch = (tid & 1) * 32;   // staging map: 64x64 tile

    // --- stage Q tile (64x64 bf16) via async path ---
    {
        const bf16_t* g = qb + ((size_t)bh * AL + qt * 64 + srow) * ADH + sch;
        bf16_t* s = &sQ[srow * ASP + sch];
#pragma unroll
        for (int c = 0; c < 4; ++c) async_copy16(s + 8 * c, g + 8 * c);
        async_wait0();
    }
    __syncthreads();

    const v16bf aq0 = load_frag(sQ + 0,  16 * w, ASP, lane);
    const v16bf aq1 = load_frag(sQ + 32, 16 * w, ASP, lane);

    v8f acc[4];
#pragma unroll
    for (int n = 0; n < 4; ++n) acc[n] = (v8f){0,0,0,0,0,0,0,0};
    float mrun[8], srun[8];
#pragma unroll
    for (int r = 0; r < 8; ++r) { mrun[r] = -1e30f; srun[r] = 0.0f; }

    for (int kt = 0; kt < AL / 64; ++kt) {
        __syncthreads();
        {   // stage K tile row-major (async), V tile transposed (manual);
            // the async K DMA overlaps the V transpose below.
            const bf16_t* gk = kb + ((size_t)bh * AL + kt * 64 + srow) * ADH + sch;
            bf16_t* sk = &sK[srow * ASP + sch];
#pragma unroll
            for (int c = 0; c < 4; ++c) async_copy16(sk + 8 * c, gk + 8 * c);

            union { uint4 q[4]; bf16_t h[32]; } vv;
            const uint4* gv = (const uint4*)(vb + ((size_t)bh * AL + kt * 64 + srow) * ADH + sch);
            vv.q[0] = gv[0]; vv.q[1] = gv[1]; vv.q[2] = gv[2]; vv.q[3] = gv[3];
#pragma unroll
            for (int e = 0; e < 32; ++e) sVt[(sch + e) * ASP + srow] = vv.h[e];
            async_wait0();
        }
        __syncthreads();

        // S = Q @ K^T  (16 rows x 64 keys per wave)
        v8f s[4];
#pragma unroll
        for (int j = 0; j < 4; ++j) {
            v16bf bk0 = load_frag(sK + 0,  16 * j, ASP, lane);
            v16bf bk1 = load_frag(sK + 32, 16 * j, ASP, lane);
            v8f c = (v8f){0,0,0,0,0,0,0,0};
            c = wmma_bf16(aq0, bk0, c);
            c = wmma_bf16(aq1, bk1, c);
            s[j] = c;
        }

        // + relative-position bias + log(padding_mask) (mask depends on col only)
        float lm[4];
#pragma unroll
        for (int j = 0; j < 4; ++j)
            lm[j] = __logf((float)pmask[bidx * AL + kt * 64 + 16 * j + l16]);
#pragma unroll
        for (int j = 0; j < 4; ++j)
#pragma unroll
            for (int r = 0; r < 8; ++r) {
                int qr = qt * 64 + 16 * w + r + 8 * half;
                int kc = kt * 64 + 16 * j + l16;
                s[j][r] += rel_emb[rel_bucket(qr, kc) * 16 + head] + lm[j];
            }

        // online softmax
        float mnew[8], scale[8];
#pragma unroll
        for (int r = 0; r < 8; ++r) {
            float mx = fmaxf(fmaxf(s[0][r], s[1][r]), fmaxf(s[2][r], s[3][r]));
            mx = redmax16(mx);
            mnew[r]  = fmaxf(mrun[r], mx);
            scale[r] = __expf(mrun[r] - mnew[r]);
            mrun[r]  = mnew[r];
        }
#pragma unroll
        for (int j = 0; j < 4; ++j)
#pragma unroll
            for (int r = 0; r < 8; ++r)
                s[j][r] = __expf(s[j][r] - mnew[r]);
#pragma unroll
        for (int r = 0; r < 8; ++r) {
            float ps = s[0][r] + s[1][r] + s[2][r] + s[3][r];
            ps = redsum16(ps);
            srun[r] = srun[r] * scale[r] + ps;
#pragma unroll
            for (int n = 0; n < 4; ++n) acc[n][r] *= scale[r];
        }

        // round-trip P through LDS to re-layout into A fragments
#pragma unroll
        for (int j = 0; j < 4; ++j)
#pragma unroll
            for (int r = 0; r < 8; ++r)
                sP[(16 * w + r + 8 * half) * ASP + 16 * j + l16] = f2bf(s[j][r]);
        // same-wave LDS RAW: DS ops in-order; compiler inserts s_wait_dscnt

        v16bf ap0 = load_frag(sP + 0,  16 * w, ASP, lane);
        v16bf ap1 = load_frag(sP + 32, 16 * w, ASP, lane);
#pragma unroll
        for (int n = 0; n < 4; ++n) {
            v16bf bv0 = load_frag(sVt + 0,  16 * n, ASP, lane);
            v16bf bv1 = load_frag(sVt + 32, 16 * n, ASP, lane);
            acc[n] = wmma_bf16(ap0, bv0, acc[n]);
            acc[n] = wmma_bf16(ap1, bv1, acc[n]);
        }
    }

    // normalize (reciprocal once per row) + store att bf16, flat [bh][l][dh]
    float rs[8];
#pragma unroll
    for (int r = 0; r < 8; ++r) rs[r] = 1.0f / srun[r];
#pragma unroll
    for (int n = 0; n < 4; ++n)
#pragma unroll
        for (int r = 0; r < 8; ++r) {
            int row = qt * 64 + 16 * w + r + 8 * half;
            attb[((size_t)bh * AL + row) * ADH + 16 * n + l16] = f2bf(acc[n][r] * rs[r]);
        }
}

// ---------------------------------------------------------------------------
// out = LayerNorm(a + res)  (no eps, biased var, no affine) ; optional bf16 copy
// One block per row of D=1024, 256 threads x 4 elems.
// ---------------------------------------------------------------------------
template <bool WB>
__global__ __launch_bounds__(256)
void add_ln_kernel(const float* __restrict__ a, const float* __restrict__ res,
                   float* __restrict__ outF, bf16_t* __restrict__ outB, int D)
{
    const int row = blockIdx.x, tid = threadIdx.x;
    __shared__ float red[256];
    const float* pa = a   + (size_t)row * D;
    const float* pr = res + (size_t)row * D;

    float v[4]; float s = 0.0f;
#pragma unroll
    for (int i = 0; i < 4; ++i) { v[i] = pa[tid + 256 * i] + pr[tid + 256 * i]; s += v[i]; }
    red[tid] = s; __syncthreads();
    for (int st = 128; st > 0; st >>= 1) { if (tid < st) red[tid] += red[tid + st]; __syncthreads(); }
    float mean = red[0] / (float)D;
    __syncthreads();

    float s2 = 0.0f;
#pragma unroll
    for (int i = 0; i < 4; ++i) { float d = v[i] - mean; s2 += d * d; }
    red[tid] = s2; __syncthreads();
    for (int st = 128; st > 0; st >>= 1) { if (tid < st) red[tid] += red[tid + st]; __syncthreads(); }
    float rinv = rsqrtf(red[0] / (float)D);

#pragma unroll
    for (int i = 0; i < 4; ++i) {
        float o = (v[i] - mean) * rinv;
        size_t idx = (size_t)row * D + tid + 256 * i;
        outF[idx] = o;
        if (WB) outB[idx] = f2bf(o);
    }
}

// ---------------------------------------------------------------------------
// Orchestration
// ---------------------------------------------------------------------------
extern "C" void kernel_launch(void* const* d_in, const int* in_sizes, int n_in,
                              void* d_out, int out_size, void* d_ws, size_t ws_size,
                              hipStream_t stream) {
    const float* x   = (const float*)d_in[0];
    const int*   pm  = (const int*)  d_in[1];
    const float* Wq  = (const float*)d_in[2];
    const float* bq  = (const float*)d_in[3];
    const float* Wk  = (const float*)d_in[4];
    const float* bk  = (const float*)d_in[5];
    const float* Wv  = (const float*)d_in[6];
    const float* bv  = (const float*)d_in[7];
    const float* Wo  = (const float*)d_in[8];
    const float* bo  = (const float*)d_in[9];
    const float* rel = (const float*)d_in[10];
    const float* W1  = (const float*)d_in[11];
    const float* b1  = (const float*)d_in[12];
    const float* W2  = (const float*)d_in[13];
    const float* b2  = (const float*)d_in[14];

    const int M = 4096;           // B*L
    const int D = 1024, F = 4096;
    const int NX = M * D;         // 4M
    const int NW = D * D;         // 1M
    const int NF = F * D;         // 4M

    char* ws = (char*)d_ws;
    const size_t MB = 1ull << 20;
    bf16_t* xb   = (bf16_t*)(ws + 0 * MB);     //  8 MB
    bf16_t* wqb  = (bf16_t*)(ws + 8 * MB);     //  2 MB
    bf16_t* wkb  = (bf16_t*)(ws + 10 * MB);
    bf16_t* wvb  = (bf16_t*)(ws + 12 * MB);
    bf16_t* wob  = (bf16_t*)(ws + 14 * MB);
    bf16_t* w1b  = (bf16_t*)(ws + 16 * MB);    //  8 MB
    bf16_t* w2b  = (bf16_t*)(ws + 24 * MB);    //  8 MB
    bf16_t* qb   = (bf16_t*)(ws + 32 * MB);    //  8 MB
    bf16_t* kbuf = (bf16_t*)(ws + 40 * MB);
    bf16_t* vbuf = (bf16_t*)(ws + 48 * MB);
    bf16_t* attb = (bf16_t*)(ws + 56 * MB);
    float*  tmp  = (float*) (ws + 64 * MB);    // 16 MB
    float*  y    = (float*) (ws + 80 * MB);    // 16 MB
    bf16_t* ybuf = (bf16_t*)(ws + 96 * MB);    //  8 MB
    bf16_t* hb   = (bf16_t*)(ws + 32 * MB);    // 32 MB, reuses qb..attb (consumed)
    float*  tmp2 = tmp;                        // reuses tmp (consumed by LN1)

    // 1) bf16 conversions (vectorized x4)
    cvt_bf16_kernel<<<NX / 4 / 256, 256, 0, stream>>>((const float4*)x,  (uint2*)xb,  NX / 4);
    cvt_bf16_kernel<<<NW / 4 / 256, 256, 0, stream>>>((const float4*)Wq, (uint2*)wqb, NW / 4);
    cvt_bf16_kernel<<<NW / 4 / 256, 256, 0, stream>>>((const float4*)Wk, (uint2*)wkb, NW / 4);
    cvt_bf16_kernel<<<NW / 4 / 256, 256, 0, stream>>>((const float4*)Wv, (uint2*)wvb, NW / 4);
    cvt_bf16_kernel<<<NW / 4 / 256, 256, 0, stream>>>((const float4*)Wo, (uint2*)wob, NW / 4);
    cvt_bf16_kernel<<<NF / 4 / 256, 256, 0, stream>>>((const float4*)W1, (uint2*)w1b, NF / 4);
    cvt_bf16_kernel<<<NF / 4 / 256, 256, 0, stream>>>((const float4*)W2, (uint2*)w2b, NF / 4);

    // 2) QKV projections (WMMA, bf16 out only)
    dim3 gD(D / GBN, M / GBM);
    gemm_bf16_kernel<0, false, true><<<gD, 256, 0, stream>>>(xb, wqb, bq, nullptr, qb,   M, D, D);
    gemm_bf16_kernel<0, false, true><<<gD, 256, 0, stream>>>(xb, wkb, bk, nullptr, kbuf, M, D, D);
    gemm_bf16_kernel<0, false, true><<<gD, 256, 0, stream>>>(xb, wvb, bv, nullptr, vbuf, M, D, D);

    // 3) fused attention (WMMA, online softmax)
    flash_attn_kernel<<<dim3(AL / 64, 64), 128, 0, stream>>>(qb, kbuf, vbuf, rel, pm, attb);

    // 4) output projection (f32 out) + residual LN (f32 + bf16 out)
    gemm_bf16_kernel<0, true, false><<<gD, 256, 0, stream>>>(attb, wob, bo, tmp, nullptr, M, D, D);
    add_ln_kernel<true><<<M, 256, 0, stream>>>(tmp, x, y, ybuf, D);

    // 5) FFN
    dim3 gF(F / GBN, M / GBM);
    gemm_bf16_kernel<1, false, true><<<gF, 256, 0, stream>>>(ybuf, w1b, b1, nullptr, hb, M, F, D);
    gemm_bf16_kernel<0, true, false><<<gD, 256, 0, stream>>>(hb, w2b, b2, tmp2, nullptr, M, D, F);
    add_ln_kernel<false><<<M, 256, 0, stream>>>(tmp2, y, (float*)d_out, nullptr, D);
}
